// SeqSelfAttention_3968549781870
// MI455X (gfx1250) — compile-verified, matrix operand order checked by reading, not run on hardware
//
#include <hip/hip_runtime.h>
#include <hip/hip_bf16.h>
#include <math.h>

typedef __attribute__((ext_vector_type(16))) __bf16 v16bf;
typedef __attribute__((ext_vector_type(8)))  float  v8f;
typedef __attribute__((ext_vector_type(4)))  unsigned int v4u;
typedef __attribute__((ext_vector_type(8)))  int    v8i;
typedef __attribute__((ext_vector_type(4)))  int    v4i;

#define B_ 4
#define C_ 128
#define T_ 1024
#define U_ 64
#define EPSV 1e-5f

#if __has_builtin(__builtin_amdgcn_tanhf)
#define FAST_TANH(x) __builtin_amdgcn_tanhf(x)
#else
#define FAST_TANH(x) tanhf(x)
#endif

// ISA 7.12.2: 16-bit A 16x32 — lanes 0-15 hold K={0..7,16..23} as pairs in
// VGPRs 0..7; lanes 16-31 hold K={8..15,24..31}. B (32x16) mirrors with M<->N.
__device__ __forceinline__ int kpair_base(int j, int hi) {
  int b = (j < 4) ? (2 * j) : (16 + 2 * (j - 4));
  return b + (hi ? 8 : 0);
}

// ---------------------------------------------------------------------------
// Kernel 1: q' = xt@Wt + bh, k = xt@Wx   (per-b GEMM M=1024,K=128,N=64)
// One wave computes a 16(t) x 64(u) tile of BOTH q' and k (shared A fragment).
// ---------------------------------------------------------------------------
__global__ __launch_bounds__(256) void qk_gemm_kernel(
    const float* __restrict__ x, const float* __restrict__ Wx,
    const float* __restrict__ Wt, const float* __restrict__ bh,
    float* __restrict__ qws, float* __restrict__ kws) {
  const int wave = (blockIdx.x * 256 + threadIdx.x) >> 5;  // 0..255
  const int lane = threadIdx.x & 31;
  const int b  = wave >> 6;          // / (T_/16)
  const int t0 = (wave & 63) << 4;
  const int L  = lane & 15;
  const int hi = lane >> 4;

  const v8f vzero = {0.f, 0.f, 0.f, 0.f, 0.f, 0.f, 0.f, 0.f};
  v8f accQ[4], accK[4];
  for (int n = 0; n < 4; ++n) { accQ[n] = vzero; accK[n] = vzero; }

  const float* xb = x + (size_t)b * C_ * T_;
  for (int kk = 0; kk < 4; ++kk) {          // K = 128 in steps of 32
    const int c0 = kk * 32;
    v16bf A;
    #pragma unroll
    for (int j = 0; j < 8; ++j) {
      const int kb = kpair_base(j, hi);
      A[2 * j]     = (__bf16)xb[(size_t)(c0 + kb)     * T_ + t0 + L];
      A[2 * j + 1] = (__bf16)xb[(size_t)(c0 + kb + 1) * T_ + t0 + L];
    }
    for (int n = 0; n < 4; ++n) {           // N = 64 in steps of 16
      const int u0 = n * 16;
      v16bf Bt, Bx;
      #pragma unroll
      for (int j = 0; j < 8; ++j) {
        const int kb = kpair_base(j, hi);
        Bt[2 * j]     = (__bf16)Wt[(size_t)(c0 + kb)     * U_ + u0 + L];
        Bt[2 * j + 1] = (__bf16)Wt[(size_t)(c0 + kb + 1) * U_ + u0 + L];
        Bx[2 * j]     = (__bf16)Wx[(size_t)(c0 + kb)     * U_ + u0 + L];
        Bx[2 * j + 1] = (__bf16)Wx[(size_t)(c0 + kb + 1) * U_ + u0 + L];
      }
      accQ[n] = __builtin_amdgcn_wmma_f32_16x16x32_bf16(
          false, A, false, Bt, (short)0, accQ[n], false, false);
      accK[n] = __builtin_amdgcn_wmma_f32_16x16x32_bf16(
          false, A, false, Bx, (short)0, accK[n], false, false);
    }
  }

  // C/D layout: lanes 0-15 rows 0..7, lanes 16-31 rows 8..15; col = lane&15.
  for (int n = 0; n < 4; ++n) {
    const int u = n * 16 + L;
    const float bhv = bh[u];
    #pragma unroll
    for (int r = 0; r < 8; ++r) {
      const int t = t0 + (hi ? 8 : 0) + r;
      const size_t idx = (size_t)(b * T_ + t) * U_ + u;
      qws[idx] = accQ[n][r] + bhv;   // fold bh into q'
      kws[idx] = accK[n][r];
    }
  }
}

// ---------------------------------------------------------------------------
// Kernel 2: one block per (b, 16-row s-tile).
//  q' tile (4KB, contiguous) staged to LDS via the Tensor Data Mover.
//  Phase A: e[s,t] = ba + sum_u tanh(q'[s,u]+k[t,u])*Wa[u]  -> out_a (L2)
//  Phase B: dense softmax per row, in place (eps denominator)
//  Phase C: v[s, :] = a[s,:] @ xt  via WMMA bf16, stored transposed (B,C,T)
// ---------------------------------------------------------------------------
__global__ __launch_bounds__(256) void attn_kernel(
    const float* __restrict__ x, const float* __restrict__ Wa,
    const float* __restrict__ ba, const float* __restrict__ qws,
    const float* __restrict__ kws, float* __restrict__ out_v,
    float* __restrict__ out_a) {
  __shared__ float4 qv4[16 * 16];   // 16 rows x 64 f32 (q' tile)
  __shared__ float4 wa4[16];        // Wa, 64 f32
  __shared__ float  red[256];

  const int tid = threadIdx.x;
  const int blk = blockIdx.x;        // 0..255
  const int b  = blk >> 6;
  const int s0 = (blk & 63) << 4;

  const float* qsrc = qws + (size_t)(b * T_ + s0) * U_;   // 1024 f32, linear

#if __has_builtin(__builtin_amdgcn_tensor_load_to_lds)
  // ---- TDM: one DMA of the whole 4KB q' tile into LDS (wave 0 issues) ----
  if (tid < 32) {
    const unsigned long long ga = (unsigned long long)(uintptr_t)qsrc;
    const unsigned int lds_off = (unsigned int)(uintptr_t)(&qv4[0]);
    // D# group0: count=1 | lds_addr | global_addr[56:0] | type=2
    v4u g0;
    g0[0] = 1u;                                   // count=1 (valid), user mode
    g0[1] = lds_off;                              // bits[63:32] lds_addr
    g0[2] = (unsigned int)(ga & 0xFFFFFFFFull);   // global_addr[31:0]
    g0[3] = (unsigned int)((ga >> 32) & 0x01FFFFFFull) | (2u << 30);  // type=2
    // D# group1: data_size=4B(code 2); tensor_dim0=tile_dim0=1024; dim1=1
    v8i g1;
    g1[0] = (int)(2u << 16);          // workgroup_mask=0, data_size=2 (4B)
    g1[1] = (int)(1024u << 16);       // tensor_dim0[15:0] in bits[63:48]
    g1[2] = (int)(1u << 16);          // tensor_dim0[31:16]=0, tensor_dim1=1
    g1[3] = (int)(1024u << 16);       // tensor_dim1 hi=0, tile_dim0=1024
    g1[4] = 1;                        // tile_dim1=1, tile_dim2=0
    g1[5] = 1024;                     // tensor_dim0_stride[31:0]
    g1[6] = 0;                        // stride hi, tensor_dim1_stride lo
    g1[7] = 0;
    const v4i gz4 = {0, 0, 0, 0};                  // groups 2/3 (<=2D tensor)
    const v8i gz8 = {0, 0, 0, 0, 0, 0, 0, 0};      // extra group (6-arg form)
    __builtin_amdgcn_tensor_load_to_lds(g0, g1, gz4, gz4, gz8, 0);
    __builtin_amdgcn_s_wait_tensorcnt(0);
  }
  for (int i = tid; i < U_ / 4; i += 256)
    wa4[i] = ((const float4*)Wa)[i];
#else
  for (int i = tid; i < 16 * 16; i += 256)
    qv4[i] = ((const float4*)qsrc)[i];
  for (int i = tid; i < U_ / 4; i += 256)
    wa4[i] = ((const float4*)Wa)[i];
#endif
  __syncthreads();

  const float bav = ba[0];
  float* arow_base = out_a + (size_t)b * T_ * T_ + (size_t)s0 * T_;

  // -------- Phase A: scores (dominant cost: 16*1024*64 tanh per block) -----
  for (int tt = 0; tt < 4; ++tt) {
    const int t = tt * 256 + tid;
    const float4* krow = (const float4*)(kws + (size_t)(b * T_ + t) * U_);
    if (tt < 3)
      __builtin_prefetch(kws + (size_t)(b * T_ + t + 256) * U_, 0, 0);
    float acc[16];
    #pragma unroll
    for (int s = 0; s < 16; ++s) acc[s] = bav;
    for (int uc = 0; uc < 16; ++uc) {          // 16 float4 chunks of u
      const float4 k4 = krow[uc];
      const float4 w4 = wa4[uc];
      #pragma unroll
      for (int s = 0; s < 16; ++s) {
        const float4 q4 = qv4[s * 16 + uc];
        acc[s] += FAST_TANH(q4.x + k4.x) * w4.x
                + FAST_TANH(q4.y + k4.y) * w4.y
                + FAST_TANH(q4.z + k4.z) * w4.z
                + FAST_TANH(q4.w + k4.w) * w4.w;
      }
    }
    #pragma unroll
    for (int s = 0; s < 16; ++s) arow_base[(size_t)s * T_ + t] = acc[s];
  }
  __syncthreads();

  // -------- Phase B: softmax per row (block reductions, float4 rows) -------
  for (int r = 0; r < 16; ++r) {
    float* row = arow_base + (size_t)r * T_;
    float4 v = *(const float4*)(row + tid * 4);
    red[tid] = fmaxf(fmaxf(v.x, v.y), fmaxf(v.z, v.w));
    __syncthreads();
    for (int off = 128; off > 0; off >>= 1) {
      if (tid < off) red[tid] = fmaxf(red[tid], red[tid + off]);
      __syncthreads();
    }
    const float m = red[0];
    __syncthreads();
    v.x = __expf(v.x - m); v.y = __expf(v.y - m);
    v.z = __expf(v.z - m); v.w = __expf(v.w - m);
    red[tid] = v.x + v.y + v.z + v.w;
    __syncthreads();
    for (int off = 128; off > 0; off >>= 1) {
      if (tid < off) red[tid] += red[tid + off];
      __syncthreads();
    }
    const float inv = 1.f / (red[0] + EPSV);
    __syncthreads();
    v.x *= inv; v.y *= inv; v.z *= inv; v.w *= inv;
    *(float4*)(row + tid * 4) = v;
  }
  __syncthreads();

  // -------- Phase C: v = a @ xt, M=16 N=128 K=1024, one wave per 16 cols ---
  const int wave = tid >> 5;
  const int lane = tid & 31;
  const int L  = lane & 15;
  const int hi = lane >> 4;
  const int c0 = wave * 16;

  const v8f vzero = {0.f, 0.f, 0.f, 0.f, 0.f, 0.f, 0.f, 0.f};
  v8f acc = vzero;
  const float* xb = x + (size_t)b * C_ * T_;
  for (int kt = 0; kt < 32; ++kt) {
    const int tb = kt * 32;
    v16bf Af, Bf;
    #pragma unroll
    for (int j = 0; j < 8; ++j) {
      const int kb = kpair_base(j, hi);   // even -> float2 aligned
      const float2 pa = *(const float2*)(arow_base + (size_t)L * T_ + tb + kb);
      const float2 pb = *(const float2*)(xb + (size_t)(c0 + L) * T_ + tb + kb);
      Af[2 * j]     = (__bf16)pa.x;
      Af[2 * j + 1] = (__bf16)pa.y;
      Bf[2 * j]     = (__bf16)pb.x;
      Bf[2 * j + 1] = (__bf16)pb.y;
    }
    acc = __builtin_amdgcn_wmma_f32_16x16x32_bf16(
        false, Af, false, Bf, (short)0, acc, false, false);
  }
  #pragma unroll
  for (int r = 0; r < 8; ++r) {
    const int s = s0 + (hi ? 8 : 0) + r;
    const int c = c0 + L;
    out_v[(size_t)(b * C_ + c) * T_ + s] = acc[r];   // (B,C,T) transposed store
  }
}

// ---------------------------------------------------------------------------
extern "C" void kernel_launch(void* const* d_in, const int* in_sizes, int n_in,
                              void* d_out, int out_size, void* d_ws,
                              size_t ws_size, hipStream_t stream) {
  const float* x  = (const float*)d_in[0];
  const float* Wx = (const float*)d_in[1];
  const float* Wt = (const float*)d_in[2];
  const float* bh = (const float*)d_in[3];
  const float* Wa = (const float*)d_in[4];
  const float* ba = (const float*)d_in[5];

  float* out_v = (float*)d_out;                          // B*C*T
  float* out_a = out_v + (size_t)B_ * C_ * T_;           // B*T*T

  float* qws = (float*)d_ws;                             // B*T*U f32
  float* kws = qws + (size_t)B_ * T_ * U_;               // B*T*U f32

  qk_gemm_kernel<<<32, 256, 0, stream>>>(x, Wx, Wt, bh, qws, kws);
  attn_kernel<<<256, 256, 0, stream>>>(x, Wa, ba, qws, kws, out_v, out_a);
}
